// GraphSAGETrain_35021163331756
// MI455X (gfx1250) — compile-verified, matrix operand order checked by reading, not run on hardware
//
#include <hip/hip_runtime.h>
#include <hip/hip_bf16.h>
#include <stdint.h>

typedef __attribute__((ext_vector_type(16))) __bf16 v16bf;
typedef __attribute__((ext_vector_type(8)))  float  v8f;
typedef __attribute__((ext_vector_type(4)))  unsigned int u32x4;
typedef __attribute__((ext_vector_type(8)))  unsigned int u32x8;

union Frag16 {
    uint4 q[2];
    v16bf v;
};

__device__ __forceinline__ unsigned short f2bf(float f) {
    union { float f; unsigned int u; } c;
    c.f = f;
    return (unsigned short)(c.u >> 16);   // truncate to bf16
}

__device__ __forceinline__ unsigned int pack2(float lo, float hi) {
    return (unsigned int)f2bf(lo) | ((unsigned int)f2bf(hi) << 16);
}

// ---------------------------------------------------------------------------
// Tensor Data Mover: 1-D DMA of `nelem` bf16 elements global -> LDS.
// D# per CDNA5 ISA ch.8: group0 {count=1, lds_addr, global_addr, type=2},
// group1 {data_size=1(2B), tensor_dim0=tile_dim0=nelem, tile_dim1=1,
//         tensor_dim0_stride=nelem}.  VADDR2/3 omitted (<=2D tensor).
// Issue from ONE wave only; tracked by TENSORcnt.
// ---------------------------------------------------------------------------
__device__ __forceinline__ void tdm_load_1d(unsigned int lds_off,
                                            const unsigned short* gptr,
                                            unsigned int nelem) {
    unsigned long long ga = (unsigned long long)(uintptr_t)gptr;
    u32x4 g0;
    g0[0] = 1u;                                         // count=1, user D#
    g0[1] = lds_off;                                    // lds_addr (bytes)
    g0[2] = (unsigned int)(ga & 0xFFFFFFFFu);           // global_addr[31:0]
    g0[3] = (unsigned int)((ga >> 32) & 0x01FFFFFFu)    // global_addr[56:32]
          | (2u << 30);                                 // type = 2 (image)
    u32x8 g1;
    g1[0] = (1u << 16);                                 // data_size=1 (2 bytes)
    g1[1] = (nelem & 0xFFFFu) << 16;                    // tensor_dim0[15:0]
    g1[2] = ((nelem >> 16) & 0xFFFFu)                   // tensor_dim0[31:16]
          | (1u << 16);                                 // tensor_dim1 = 1
    g1[3] = (nelem & 0xFFFFu) << 16;                    // tile_dim0 = nelem
    g1[4] = 1u;                                         // tile_dim1 = 1
    g1[5] = nelem;                                      // tensor_dim0_stride
    g1[6] = 0u;
    g1[7] = 0u;
    asm volatile("tensor_load_to_lds %0, %1" :: "s"(g0), "s"(g1) : "memory");
}

// ---------------------------------------------------------------------------
// Zero fill
// ---------------------------------------------------------------------------
__global__ void zero_kernel(float* __restrict__ p, long long n) {
    long long i = (long long)blockIdx.x * blockDim.x + threadIdx.x;
    long long stride = (long long)gridDim.x * blockDim.x;
    for (; i < n; i += stride) p[i] = 0.0f;
}

// ---------------------------------------------------------------------------
// Weight pre-convert: fp32 [K, HOUT] -> bf16 chunk-major [K/32][HOUT][32]
// Each 16KB (HOUT=256) chunk is a contiguous 1-D block for the TDM.
// ---------------------------------------------------------------------------
__global__ void convert_w_kernel(const float* __restrict__ W,
                                 unsigned short* __restrict__ Wt,
                                 int K, int HOUT) {
    long long total = (long long)K * HOUT;
    for (long long i = (long long)blockIdx.x * blockDim.x + threadIdx.x;
         i < total; i += (long long)gridDim.x * blockDim.x) {
        int k = (int)(i / HOUT);
        int n = (int)(i - (long long)k * HOUT);
        Wt[(long long)(k >> 5) * HOUT * 32 + n * 32 + (k & 31)] = f2bf(W[i]);
    }
}

// ---------------------------------------------------------------------------
// Edge scatter-add: one wave per edge.  Gathers hit the 192MB L2 (feature
// tables fit); atomics resolve in L2 via hardware global_atomic_add_f32.
// ---------------------------------------------------------------------------
template<int D>
__global__ void scatter_kernel(const float* __restrict__ h,
                               const int* __restrict__ src,
                               const int* __restrict__ dst,
                               float* __restrict__ agg,
                               int n_edges) {
    int wave = threadIdx.x >> 5;
    int lane = threadIdx.x & 31;
    int e = blockIdx.x * 8 + wave;
    if (e >= n_edges) return;
    int s = src[e];
    int d = dst[e];
    const float4* hp = (const float4*)(h + (long long)s * D);
    float* ap = agg + (long long)d * D;
#pragma unroll
    for (int c = 0; c < D / 128; ++c) {
        float4 v = hp[lane + c * 32];
        int base = (lane + c * 32) * 4;
        unsafeAtomicAdd(ap + base + 0, v.x);
        unsafeAtomicAdd(ap + base + 1, v.y);
        unsafeAtomicAdd(ap + base + 2, v.z);
        unsafeAtomicAdd(ap + base + 3, v.w);
    }
}

// ---------------------------------------------------------------------------
// Node update:  out = relu( concat(self, (agg-self)*norm) @ W + b )
//   K = 2*DIN, HOUT = 256.  Block: 256 thr = 8 waves, 32 rows x 256 cols.
//   B tiles DMA'd into double-buffered LDS by the TDM (overlaps WMMA).
//   Epilogue staged through LDS for coalesced b128 stores.
// ---------------------------------------------------------------------------
template<int DIN, int HOUT>
__global__ void layer_kernel(const float* __restrict__ h_in,
                             const int* __restrict__ map,
                             const float* __restrict__ agg,
                             const float* __restrict__ norm,
                             const unsigned short* __restrict__ Wt, // bf16 chunks
                             const float* __restrict__ b,           // [HOUT]
                             float* __restrict__ out,               // [n_out, HOUT]
                             int n_out) {
    constexpr int K = 2 * DIN;
    constexpr int NCHUNK = K / 32;
    constexpr int CHUNK = HOUT * 32;            // elements per B chunk
    __shared__ __align__(16) unsigned char smem[32 * 32 * 2 + 2 * CHUNK * 2];
    unsigned short* lA = (unsigned short*)smem;                      // 2KB
    unsigned short* lB = (unsigned short*)(smem + 32 * 32 * 2);      // 2x16KB
    float* stage = (float*)(smem + 32 * 32 * 2);  // 32xHOUT f32, aliases lB
    __shared__ int   sMap[32];
    __shared__ float sNorm[32];

    const int tid = threadIdx.x;
    const int block_row = blockIdx.x * 32;

    if (tid < 32) {
        int r = block_row + tid;
        if (r >= n_out) r = n_out - 1;
        sMap[tid]  = map[r];
        sNorm[tid] = norm[r];
        __builtin_prefetch(&agg[(long long)r * DIN], 0, 1);
    }

    const int wave = tid >> 5;
    const int lane = tid & 31;
    const int mrow = (wave & 1) * 16;
    const int ncol = (wave >> 1) * 64;

    // kick off DMA of B chunk 0
    if (wave == 0) {
        tdm_load_1d((unsigned int)(uintptr_t)lB, Wt, CHUNK);
    }
    __syncthreads();

    const v8f vzero = {0.f, 0.f, 0.f, 0.f, 0.f, 0.f, 0.f, 0.f};
    v8f acc[4] = {vzero, vzero, vzero, vzero};

    // thread -> (row, 4 k-values) for A staging: 256 thr * 4 = 1024 = 32x32
    const int am = tid >> 3;
    const int ak = (tid & 7) * 4;

    for (int c = 0; c < NCHUNK; ++c) {
        const int k0 = c * 32;
        // ---- stage A: each thread builds 4 bf16 values, one ds_store_b64 ----
        {
            int gk = k0 + ak;                       // chunk is all-self or all-diff
            bool is_self = (gk < DIN);
            int fd = is_self ? gk : gk - DIN;
            float4 sv = *(const float4*)&h_in[(long long)sMap[am] * DIN + fd];
            float4 v;
            if (is_self) {
                v = sv;
            } else {
                int row = block_row + am;
                if (row >= n_out) row = n_out - 1;
                float4 a4 = *(const float4*)&agg[(long long)row * DIN + fd];
                float nm = sNorm[am];
                v.x = (a4.x - sv.x) * nm;
                v.y = (a4.y - sv.y) * nm;
                v.z = (a4.z - sv.z) * nm;
                v.w = (a4.w - sv.w) * nm;
            }
            uint2 packed;
            packed.x = pack2(v.x, v.y);
            packed.y = pack2(v.z, v.w);
            *(uint2*)&lA[am * 32 + ak] = packed;
        }
        // ---- prefetch-next B chunk via TDM; wait for current chunk ----
        if (wave == 0) {
            if (c + 1 < NCHUNK) {
                tdm_load_1d((unsigned int)(uintptr_t)(lB + ((c + 1) & 1) * CHUNK),
                            Wt + (long long)(c + 1) * CHUNK, CHUNK);
                __builtin_amdgcn_s_wait_tensorcnt(1);   // chunk c landed
            } else {
                __builtin_amdgcn_s_wait_tensorcnt(0);
            }
        }
        __syncthreads();

        // ---- fragments + WMMA (all B frags loaded up front) ----
        const unsigned short* curB = lB + (c & 1) * CHUNK;
        const int kb = (lane >> 4) * 8;
        Frag16 a;
        {
            int m = mrow + (lane & 15);
            a.q[0] = *(const uint4*)&lA[m * 32 + kb];
            a.q[1] = *(const uint4*)&lA[m * 32 + 16 + kb];
        }
        Frag16 bf[4];
#pragma unroll
        for (int s = 0; s < 4; ++s) {
            int n = ncol + s * 16 + (lane & 15);
            bf[s].q[0] = *(const uint4*)&curB[n * 32 + kb];
            bf[s].q[1] = *(const uint4*)&curB[n * 32 + 16 + kb];
        }
#pragma unroll
        for (int s = 0; s < 4; ++s) {
            acc[s] = __builtin_amdgcn_wmma_f32_16x16x32_bf16(
                false, a.v, false, bf[s].v, (short)0, acc[s], false, false);
        }
        __syncthreads();
    }

    // ---- epilogue: dump C tile to LDS, then coalesced bias+relu stores ----
#pragma unroll
    for (int s = 0; s < 4; ++s) {
        int col = ncol + s * 16 + (lane & 15);
        int rb = mrow + 8 * (lane >> 4);
#pragma unroll
        for (int r = 0; r < 8; ++r) {
            stage[(rb + r) * HOUT + col] = acc[s][r];
        }
    }
    __syncthreads();

    const int orow = tid >> 3;              // 0..31
    const int ocol = (tid & 7) * 32;        // 32-col strip
    if (block_row + orow < n_out) {
        long long gbase = (long long)(block_row + orow) * HOUT + ocol;
#pragma unroll
        for (int j = 0; j < 8; ++j) {
            float4 v  = *(const float4*)&stage[orow * HOUT + ocol + j * 4];
            float4 bb = *(const float4*)&b[ocol + j * 4];
            v.x = fmaxf(v.x + bb.x, 0.f);
            v.y = fmaxf(v.y + bb.y, 0.f);
            v.z = fmaxf(v.z + bb.z, 0.f);
            v.w = fmaxf(v.w + bb.w, 0.f);
            *(float4*)&out[gbase + j * 4] = v;
        }
    }
}

// ---------------------------------------------------------------------------
// FC: out[N2,64] = h2[N2,256] @ Wfc[256,64] + bfc.  Block: 256 thr = 8 waves,
// 128 rows per block.  Whole Wfc (bf16 chunk-major [8][64][32]) DMA'd into
// LDS once by the TDM.  Epilogue staged through LDS.
// ---------------------------------------------------------------------------
__global__ void fc_kernel(const float* __restrict__ h2,
                          const unsigned short* __restrict__ Wt,  // 16384 bf16
                          const float* __restrict__ bfc,          // [64]
                          float* __restrict__ out,                // [n_rows, 64]
                          int n_rows) {
    __shared__ __align__(16) unsigned char smem[64 * 256 * 2 + 128 * 32 * 2];
    unsigned short* lB = (unsigned short*)smem;                    // 32KB
    unsigned short* lA = (unsigned short*)(smem + 64 * 256 * 2);   // 8KB
    float* stage = (float*)smem;            // 128x64 f32 = 32KB, aliases lB

    const int tid = threadIdx.x;
    const int wave = tid >> 5;
    const int lane = tid & 31;
    const int mrow = wave * 16;
    const int block_row = blockIdx.x * 128;

    if (wave == 0) {
        tdm_load_1d((unsigned int)(uintptr_t)lB, Wt, 64 * 256);
        __builtin_amdgcn_s_wait_tensorcnt(0);
    }

    const v8f vzero = {0.f, 0.f, 0.f, 0.f, 0.f, 0.f, 0.f, 0.f};
    v8f acc[4] = {vzero, vzero, vzero, vzero};

    // A staging: 128x32 tile, each thread packs 16 values
    const int am = tid >> 1;            // 0..127
    const int ak = (tid & 1) * 16;      // 0 or 16

    for (int c = 0; c < 8; ++c) {
        const int k0 = c * 32;
        __syncthreads();
        {
            int row = block_row + am;
            if (row >= n_rows) row = n_rows - 1;
            const float* srcp = &h2[(long long)row * 256 + k0 + ak];
#pragma unroll
            for (int j = 0; j < 4; ++j) {
                float4 v = *(const float4*)(srcp + j * 4);
                uint2 packed;
                packed.x = pack2(v.x, v.y);
                packed.y = pack2(v.z, v.w);
                *(uint2*)&lA[am * 32 + ak + j * 4] = packed;
            }
        }
        __syncthreads();

        const int kb = (lane >> 4) * 8;
        Frag16 a;
        {
            int m = mrow + (lane & 15);
            a.q[0] = *(const uint4*)&lA[m * 32 + kb];
            a.q[1] = *(const uint4*)&lA[m * 32 + 16 + kb];
        }
        const unsigned short* curB = &lB[c * 64 * 32];
        Frag16 bf[4];
#pragma unroll
        for (int s = 0; s < 4; ++s) {
            int n = s * 16 + (lane & 15);
            bf[s].q[0] = *(const uint4*)&curB[n * 32 + kb];
            bf[s].q[1] = *(const uint4*)&curB[n * 32 + 16 + kb];
        }
#pragma unroll
        for (int s = 0; s < 4; ++s) {
            acc[s] = __builtin_amdgcn_wmma_f32_16x16x32_bf16(
                false, a.v, false, bf[s].v, (short)0, acc[s], false, false);
        }
    }

    // ---- epilogue through LDS ----
    __syncthreads();                         // done reading lB
#pragma unroll
    for (int s = 0; s < 4; ++s) {
        int col = s * 16 + (lane & 15);
        int rb = mrow + 8 * (lane >> 4);
#pragma unroll
        for (int r = 0; r < 8; ++r) {
            stage[(rb + r) * 64 + col] = acc[s][r];
        }
    }
    __syncthreads();

    const int orow = tid >> 1;               // 0..127
    const int ocol = (tid & 1) * 32;         // 0 or 32
    if (block_row + orow < n_rows) {
        long long gbase = (long long)(block_row + orow) * 64 + ocol;
#pragma unroll
        for (int j = 0; j < 8; ++j) {
            float4 v  = *(const float4*)&stage[orow * 64 + ocol + j * 4];
            float4 bb = *(const float4*)&bfc[ocol + j * 4];
            v.x += bb.x; v.y += bb.y; v.z += bb.z; v.w += bb.w;
            *(float4*)&out[gbase + j * 4] = v;
        }
    }
}

// ---------------------------------------------------------------------------
// Host-side launch
// ---------------------------------------------------------------------------
extern "C" void kernel_launch(void* const* d_in, const int* in_sizes, int n_in,
                              void* d_out, int out_size, void* d_ws, size_t ws_size,
                              hipStream_t stream) {
    const float* x     = (const float*)d_in[0];
    const int*   map1  = (const int*)  d_in[1];
    const int*   map2  = (const int*)  d_in[2];
    const int*   src0  = (const int*)  d_in[3];
    const int*   dst0  = (const int*)  d_in[4];
    const int*   src1  = (const int*)  d_in[5];
    const int*   dst1  = (const int*)  d_in[6];
    const float* norm1 = (const float*)d_in[7];
    const float* norm2 = (const float*)d_in[8];
    const float* W1    = (const float*)d_in[9];
    const float* b1    = (const float*)d_in[10];
    const float* W2    = (const float*)d_in[11];
    const float* b2    = (const float*)d_in[12];
    const float* Wfc   = (const float*)d_in[13];
    const float* bfc   = (const float*)d_in[14];
    float* out = (float*)d_out;

    const int N1 = in_sizes[1];
    const int N2 = in_sizes[2];
    const int E0 = in_sizes[3];
    const int E1 = in_sizes[5];
    const int D = 128, H = 256;

    float* agg1 = (float*)d_ws;                       // N1 * 128 f32
    float* h1   = agg1 + (long long)N1 * D;           // N1 * 256 f32
    float* agg2 = h1   + (long long)N1 * H;           // N2 * 256 f32
    float* h2   = agg2 + (long long)N2 * H;           // N2 * 256 f32
    unsigned short* W1t  = (unsigned short*)(h2 + (long long)N2 * H); // 256*256
    unsigned short* W2t  = W1t + 256 * 256;                           // 512*256
    unsigned short* Wfct = W2t + 512 * 256;                           // 256*64

    // weight pre-convert (tiny) + zero aggregation buffers
    convert_w_kernel<<<64,  256, 0, stream>>>(W1,  W1t,  256, 256);
    convert_w_kernel<<<128, 256, 0, stream>>>(W2,  W2t,  512, 256);
    convert_w_kernel<<<16,  256, 0, stream>>>(Wfc, Wfct, 256, 64);
    zero_kernel<<<4096, 256, 0, stream>>>(agg1, (long long)N1 * D);
    zero_kernel<<<4096, 256, 0, stream>>>(agg2, (long long)N2 * H);

    // ---- layer 1 ----
    scatter_kernel<128><<<(E0 + 7) / 8, 256, 0, stream>>>(x, src0, dst0, agg1, E0);
    layer_kernel<128, 256><<<(N1 + 31) / 32, 256, 0, stream>>>(
        x, map1, agg1, norm1, W1t, b1, h1, N1);

    // ---- layer 2 ----
    scatter_kernel<256><<<(E1 + 7) / 8, 256, 0, stream>>>(h1, src1, dst1, agg2, E1);
    layer_kernel<256, 256><<<(N2 + 31) / 32, 256, 0, stream>>>(
        h1, map2, agg2, norm2, W2t, b2, h2, N2);

    // ---- FC ----
    fc_kernel<<<(N2 + 127) / 128, 256, 0, stream>>>(h2, Wfct, bfc, out, N2);
}
